// EncoderLayer_87780541596348
// MI455X (gfx1250) — compile-verified, matrix operand order checked by reading, not run on hardware
//
#include <hip/hip_runtime.h>
#include <hip/hip_bf16.h>
#include <math.h>

typedef _Float16 half_t;
typedef __attribute__((ext_vector_type(16))) _Float16 v16h;
typedef __attribute__((ext_vector_type(8)))  _Float16 v8h;
typedef __attribute__((ext_vector_type(8)))  float    v8f;

#define D_MODEL 1024
#define NHEAD   16
#define D_HEAD  64
#define D_FF    4096
#define BATCH   2
#define SEQ     2048
#define NTOK    (BATCH * SEQ)      /* 4096 rows */
#define EPS     1e-5f

// --- TDM availability (arity differs between toolchains) --------------------
#if defined(__has_builtin)
#  if __has_builtin(__builtin_amdgcn_tensor_load_to_lds)
#    define HAVE_TDM 1
#  endif
#endif
#ifndef HAVE_TDM
#  define HAVE_TDM 0
#endif

static __device__ inline v8f vzero8f() {
    v8f z;
#pragma unroll
    for (int i = 0; i < 8; ++i) z[i] = 0.0f;
    return z;
}

// ---------------------------------------------------------------------------
// CDNA5 async global->LDS copy (ASYNCcnt-tracked), 16 bytes per lane.
// ---------------------------------------------------------------------------
static __device__ inline void async_copy_b128(void* lds_dst, const void* gsrc) {
    unsigned l = (unsigned)(uintptr_t)lds_dst;     // LDS byte offset
    asm volatile("global_load_async_to_lds_b128 %0, %1, off"
                 :: "v"(l), "v"(gsrc)
                 : "memory");
}
#define WAIT_ASYNC(n) asm volatile("s_wait_asynccnt " #n ::: "memory")

// ---------------------------------------------------------------------------
// CDNA5 Tensor Data Mover: 2-D tile load Global->LDS (TENSORcnt-tracked).
// Descriptor per CDNA5 ISA. Issued once per wave;
// EXEC is ignored by TDM, so call it from a single wave per block.
// ---------------------------------------------------------------------------
#if HAVE_TDM
typedef __attribute__((ext_vector_type(4))) unsigned int u32x4;
typedef __attribute__((ext_vector_type(8))) int          i32x8;
typedef __attribute__((ext_vector_type(4))) int          i32x4;

static __device__ inline void tdm_load_2d(void* lds, const void* g,
                                          unsigned dim0, unsigned dim1,
                                          unsigned tile0, unsigned tile1,
                                          unsigned stride0 /* elements */) {
    unsigned long long ga = (unsigned long long)(uintptr_t)g;
    unsigned           lo = (unsigned)(uintptr_t)lds;
    u32x4 g0;
    g0[0] = 1u;                                   // count=1 (valid), user mode
    g0[1] = lo;                                   // lds_addr (bytes)
    g0[2] = (unsigned)ga;                         // global_addr[31:0]
    g0[3] = ((unsigned)(ga >> 32) & 0x01FFFFFFu)  // global_addr[56:32]
            | (2u << 30);                         // type = 2 ("image")
    i32x8 g1;
    g1[0] = 0x00010000;                           // wg_mask=0, data_size=1 (2B)
    g1[1] = (int)((dim0 & 0xFFFFu) << 16);        // tensor_dim0[15:0] @ bit48
    g1[2] = (int)((dim0 >> 16) | ((dim1 & 0xFFFFu) << 16)); // dim0 hi | dim1 lo
    g1[3] = (int)((dim1 >> 16) | (tile0 << 16));  // dim1 hi | tile_dim0
    g1[4] = (int)(tile1 & 0xFFFFu);               // tile_dim1 | tile_dim2=0
    g1[5] = (int)stride0;                         // tensor_dim0_stride[31:0]
    g1[6] = 0;                                    // stride0 hi | stride1 lo
    g1[7] = 0;                                    // stride1 hi
    i32x4 z4; z4[0] = z4[1] = z4[2] = z4[3] = 0;  // groups 2/3 unused (2-D)
#if __clang_major__ >= 23
    i32x8 z8;
#pragma unroll
    for (int i = 0; i < 8; ++i) z8[i] = 0;
    __builtin_amdgcn_tensor_load_to_lds(g0, g1, z4, z4, z8, 0);
#else
    __builtin_amdgcn_tensor_load_to_lds(g0, g1, z4, z4, 0);
#endif
}
#endif

// Load one 16x32 f16 WMMA fragment from an LDS tile laid out [row][k]
// (row stride = `stride` halves), per the CDNA5 16-bit A/B layout:
// lanes 0-15 hold K = {k0..k0+7, k0+16..k0+23} of row (lane&15),
// lanes 16-31 hold K = {k0+8..k0+15, k0+24..k0+31}.
static __device__ inline v16h frag_ld(const half_t* tile, int stride, int rowbase, int k0) {
    int lane = threadIdx.x & 31;
    int r    = lane & 15;
    int hi   = lane >> 4;
    const half_t* rp = tile + (rowbase + r) * stride + k0 + hi * 8;
    v8h lo  = *(const v8h*)(rp);
    v8h hi8 = *(const v8h*)(rp + 16);
    return __builtin_shufflevector(lo, hi8, 0,1,2,3,4,5,6,7,8,9,10,11,12,13,14,15);
}

// ---------------------------------------------------------------------------
// fp32 -> f16 conversion
// ---------------------------------------------------------------------------
__global__ __launch_bounds__(256) void cvt_f16_kernel(const float* __restrict__ in,
                                                      half_t* __restrict__ out, int n) {
    int i = blockIdx.x * 256 + threadIdx.x;
    if (i < n) out[i] = (half_t)in[i];
}

// ---------------------------------------------------------------------------
// GEMM:  out[M,N] = A[M,K] @ W[N,K]^T + bias[N]   (optional relu)
// Block: 256 threads = 8 waves; tile 128(M) x 128(N), K staged 32 at a time.
// A tile: per-lane async global->LDS copies (ASYNCcnt).
// B (weight) tile: one TDM descriptor issued by wave 0 (TENSORcnt).
// Double-buffered. Wave w -> rows (w&3)*32, cols (w>>2)*64: 8 WMMAs/K-step.
// ---------------------------------------------------------------------------
#define TM 128
#define TN 128
#define TK 32

__global__ __launch_bounds__(256) void gemm_bias_kernel(
    const half_t* __restrict__ A, const half_t* __restrict__ W,
    const float* __restrict__ bias, float* __restrict__ outF,
    half_t* __restrict__ outH, int M, int N, int K, int relu)
{
    __shared__ __align__(16) half_t As[2][TM * TK];   // 2 x 8 KB
    __shared__ __align__(16) half_t Bs[2][TN * TK];   // 2 x 8 KB

    const int tid  = threadIdx.x;
    const int wave = tid >> 5;
    const int lane = tid & 31;
    const int m0 = blockIdx.y * TM;
    const int n0 = blockIdx.x * TN;
    const int wr = (wave & 3) * 32;   // wave row base within tile (2 subtiles)
    const int wc = (wave >> 2) * 64;  // wave col base within tile (4 subtiles)

    v8f acc[8];
#pragma unroll
    for (int s = 0; s < 8; ++s) acc[s] = vzero8f();

    // Stage A tile 128x32: 2 async b128 per thread.
    auto stage_A = [&](int buf, int k0) {
#pragma unroll
        for (int t = 0; t < 2; ++t) {
            int idx = tid + t * 256;          // 0..511
            int r = idx >> 2;                 // 0..127
            int c = (idx & 3) * 8;            // 0,8,16,24
            async_copy_b128(&As[buf][r * TK + c],
                            A + (size_t)(m0 + r) * K + k0 + c);
        }
    };
    // Stage B tile 128x32 (rows n0..n0+127 of W[N,K]).
    auto stage_B = [&](int buf, int k0) {
#if HAVE_TDM
        if (wave == 0)
            tdm_load_2d(&Bs[buf][0], W + (size_t)n0 * K + k0,
                        (unsigned)K, (unsigned)N, TK, TN, (unsigned)K);
#else
#pragma unroll
        for (int t = 0; t < 2; ++t) {
            int idx = tid + t * 256;
            int r = idx >> 2;
            int c = (idx & 3) * 8;
            async_copy_b128(&Bs[buf][r * TK + c],
                            W + (size_t)(n0 + r) * K + k0 + c);
        }
#endif
    };

    const int nk = K / TK;
    stage_A(0, 0);
    stage_B(0, 0);
    for (int ki = 0; ki < nk; ++ki) {
        const int cur = ki & 1;
        if (ki + 1 < nk) {
            stage_A(cur ^ 1, (ki + 1) * TK);  // prefetch next tile
            stage_B(cur ^ 1, (ki + 1) * TK);
#if HAVE_TDM
            WAIT_ASYNC(2);                    // tile-ki A copies (this wave)
            __builtin_amdgcn_s_wait_tensorcnt((short)1);  // tile-ki TDM done
#else
            WAIT_ASYNC(4);
#endif
        } else {
            WAIT_ASYNC(0);
#if HAVE_TDM
            __builtin_amdgcn_s_wait_tensorcnt((short)0);
#endif
        }
        __syncthreads();                      // all waves' tile-ki data visible

        v16h a0 = frag_ld(As[cur], TK, wr, 0);
        v16h a1 = frag_ld(As[cur], TK, wr + 16, 0);
#pragma unroll
        for (int s = 0; s < 4; ++s) {
            v16h b = frag_ld(Bs[cur], TK, wc + s * 16, 0);
            acc[s] = __builtin_amdgcn_wmma_f32_16x16x32_f16(
                false, a0, false, b, (short)0, acc[s], false, false);
            acc[s + 4] = __builtin_amdgcn_wmma_f32_16x16x32_f16(
                false, a1, false, b, (short)0, acc[s + 4], false, false);
        }
        __syncthreads();                      // done reading buf before reuse
    }

    // Epilogue: bias (+relu), store f32 and/or f16.
    const int hi  = lane >> 4;
    const int col = lane & 15;
#pragma unroll
    for (int t = 0; t < 2; ++t)
#pragma unroll
        for (int s = 0; s < 4; ++s) {
            int n = n0 + wc + s * 16 + col;
            float bv = bias ? bias[n] : 0.0f;
#pragma unroll
            for (int r = 0; r < 8; ++r) {
                int m = m0 + wr + t * 16 + r + hi * 8;
                float v = acc[t * 4 + s][r] + bv;
                if (relu) v = fmaxf(v, 0.0f);
                if (outF) outF[(size_t)m * N + n] = v;
                if (outH) outH[(size_t)m * N + n] = (half_t)v;
            }
        }
}

// ---------------------------------------------------------------------------
// Flash attention (per batch*head, 64-query tile), d_head = 64.
// Q/K/V/O are [NTOK, 1024] f16, head h occupies columns h*64..h*64+63.
// Block = 128 threads (4 waves). Wave w owns output rows w*16..w*16+15.
// Q/K tiles staged via async global->LDS; V transposed via ds scatter.
// ---------------------------------------------------------------------------
__global__ __launch_bounds__(128) void flash_attn_kernel(
    const half_t* __restrict__ Q, const half_t* __restrict__ Km,
    const half_t* __restrict__ V, half_t* __restrict__ O)
{
    __shared__ __align__(16) half_t Qs[64 * 64];     // 8 KB
    __shared__ __align__(16) half_t Ks[64 * 64];     // 8 KB
    __shared__ __align__(16) half_t VsT[64 * 64];    // 8 KB  [d][key]
    __shared__ __align__(16) half_t Ps[64 * 64];     // 8 KB
    __shared__ __align__(16) float  Ss[64 * 64];     // 16 KB
    __shared__ float corr_s[64];
    __shared__ float lsum_s[64];

    const int tid  = threadIdx.x;
    const int wave = tid >> 5;
    const int lane = tid & 31;
    const int hi   = lane >> 4;
    const int col  = lane & 15;

    const int bh = blockIdx.y;
    const int b  = bh >> 4;
    const int h  = bh & 15;
    const int q0 = blockIdx.x * 64;
    const size_t headoff = (size_t)h * D_HEAD;
    const size_t rowbase = (size_t)b * SEQ;

    // Stage Q tile (64 rows x 64 halves) asynchronously.
    for (int t = tid; t < 512; t += 128) {
        int r = t >> 3, c = (t & 7) * 8;
        async_copy_b128(&Qs[r * 64 + c],
                        Q + (rowbase + q0 + r) * D_MODEL + headoff + c);
    }

    float m_run = -1e30f, l_run = 0.0f;   // valid for tid < 64 (row owner)
    v8f o[4];
#pragma unroll
    for (int s = 0; s < 4; ++s) o[s] = vzero8f();

    const float scale = 0.125f;           // 1/sqrt(64)

    for (int j = 0; j < SEQ; j += 64) {
        // Stage K tile [key][d] (async) and V tile transposed [d][key].
        for (int t = tid; t < 512; t += 128) {
            int r = t >> 3, c = (t & 7) * 8;
            async_copy_b128(&Ks[r * 64 + c],
                            Km + (rowbase + j + r) * D_MODEL + headoff + c);
            v8h vv = *(const v8h*)(V + (rowbase + j + r) * D_MODEL + headoff + c);
#pragma unroll
            for (int i = 0; i < 8; ++i) VsT[(c + i) * 64 + r] = vv[i];
        }
        WAIT_ASYNC(0);
        __syncthreads();

        // S = Q @ K^T  (64x64), each wave: 4 col subtiles, K-dim = 64 (2 steps)
        v8f sa[4];
#pragma unroll
        for (int s = 0; s < 4; ++s) sa[s] = vzero8f();
#pragma unroll
        for (int kk = 0; kk < 64; kk += 32) {
            v16h a = frag_ld(Qs, 64, wave * 16, kk);
#pragma unroll
            for (int s = 0; s < 4; ++s) {
                v16h bfr = frag_ld(Ks, 64, s * 16, kk);
                sa[s] = __builtin_amdgcn_wmma_f32_16x16x32_f16(
                    false, a, false, bfr, (short)0, sa[s], false, false);
            }
        }
        // Spill S tile to LDS (fp32).
#pragma unroll
        for (int s = 0; s < 4; ++s)
#pragma unroll
            for (int r = 0; r < 8; ++r)
                Ss[(wave * 16 + r + hi * 8) * 64 + s * 16 + col] = sa[s][r];
        __syncthreads();

        // Online softmax: threads 0..63 each own one query row.
        if (tid < 64) {
            const float* srow = Ss + tid * 64;
            float mnew = m_run;
            for (int c = 0; c < 64; ++c) {
                float v = srow[c] * scale;
                mnew = fmaxf(mnew, v);
            }
            float corr = __expf(m_run - mnew);
            float lacc = l_run * corr;
            half_t* prow = Ps + tid * 64;
            for (int c = 0; c < 64; ++c) {
                float p = __expf(srow[c] * scale - mnew);
                lacc += p;
                prow[c] = (half_t)p;
            }
            m_run = mnew;
            l_run = lacc;
            corr_s[tid] = corr;
        }
        __syncthreads();

        // Rescale O accumulators, then O += P @ V.
#pragma unroll
        for (int s = 0; s < 4; ++s)
#pragma unroll
            for (int r = 0; r < 8; ++r)
                o[s][r] *= corr_s[wave * 16 + r + hi * 8];
#pragma unroll
        for (int kk = 0; kk < 64; kk += 32) {
            v16h a = frag_ld(Ps, 64, wave * 16, kk);
#pragma unroll
            for (int s = 0; s < 4; ++s) {
                v16h bfr = frag_ld(VsT, 64, s * 16, kk);
                o[s] = __builtin_amdgcn_wmma_f32_16x16x32_f16(
                    false, a, false, bfr, (short)0, o[s], false, false);
            }
        }
        __syncthreads();
    }

    if (tid < 64) lsum_s[tid] = l_run;
    __syncthreads();

    // Normalize and store O tile (f16).
#pragma unroll
    for (int s = 0; s < 4; ++s)
#pragma unroll
        for (int r = 0; r < 8; ++r) {
            int row = wave * 16 + r + hi * 8;
            float v = o[s][r] / lsum_s[row];
            O[(rowbase + q0 + row) * D_MODEL + headoff + s * 16 + col] = (half_t)v;
        }
}

// ---------------------------------------------------------------------------
// Residual add + LayerNorm, one block (256 threads) per row of 1024.
// outF (f32, may be null) and outH (f16, may be null).
// ---------------------------------------------------------------------------
__global__ __launch_bounds__(256) void residual_ln_kernel(
    const float* __restrict__ x, const float* __restrict__ y,
    const float* __restrict__ gamma, const float* __restrict__ beta,
    float* __restrict__ outF, half_t* __restrict__ outH)
{
    __shared__ float sA[256];
    __shared__ float sB[256];
    const int row = blockIdx.x;
    const int tid = threadIdx.x;
    const size_t base = (size_t)row * D_MODEL;

    float v[4];
    float s1 = 0.0f, s2 = 0.0f;
#pragma unroll
    for (int i = 0; i < 4; ++i) {
        int c = tid + i * 256;
        v[i] = x[base + c] + y[base + c];
        s1 += v[i];
        s2 += v[i] * v[i];
    }
    sA[tid] = s1; sB[tid] = s2;
    __syncthreads();
    for (int stride = 128; stride > 0; stride >>= 1) {
        if (tid < stride) { sA[tid] += sA[tid + stride]; sB[tid] += sB[tid + stride]; }
        __syncthreads();
    }
    const float mu  = sA[0] * (1.0f / D_MODEL);
    const float var = sB[0] * (1.0f / D_MODEL) - mu * mu;
    const float inv = rsqrtf(var + EPS);
#pragma unroll
    for (int i = 0; i < 4; ++i) {
        int c = tid + i * 256;
        float o = (v[i] - mu) * inv * gamma[c] + beta[c];
        if (outF) outF[base + c] = o;
        if (outH) outH[base + c] = (half_t)o;
    }
}

// ---------------------------------------------------------------------------
// Host-side orchestration
// ---------------------------------------------------------------------------
extern "C" void kernel_launch(void* const* d_in, const int* in_sizes, int n_in,
                              void* d_out, int out_size, void* d_ws, size_t ws_size,
                              hipStream_t stream) {
    const float* x   = (const float*)d_in[0];
    const float* Wq  = (const float*)d_in[1];  const float* bq  = (const float*)d_in[2];
    const float* Wk  = (const float*)d_in[3];  const float* bk  = (const float*)d_in[4];
    const float* Wv  = (const float*)d_in[5];  const float* bv  = (const float*)d_in[6];
    const float* Wo  = (const float*)d_in[7];  const float* bo  = (const float*)d_in[8];
    const float* W1  = (const float*)d_in[9];  const float* b1  = (const float*)d_in[10];
    const float* W2  = (const float*)d_in[11]; const float* b2  = (const float*)d_in[12];
    const float* g1  = (const float*)d_in[13]; const float* be1 = (const float*)d_in[14];
    const float* g2  = (const float*)d_in[15]; const float* be2 = (const float*)d_in[16];
    float* out = (float*)d_out;

    char* ws = (char*)d_ws;
    size_t off = 0;
    auto alloc = [&](size_t bytes) -> void* {
        void* p = ws + off;
        off = (off + bytes + 255) & ~(size_t)255;
        return p;
    };

    const size_t NE_X  = (size_t)NTOK * D_MODEL;    // 4M
    const size_t NE_W  = (size_t)D_MODEL * D_MODEL; // 1M
    const size_t NE_W1 = (size_t)D_FF * D_MODEL;    // 4M
    const size_t NE_F  = (size_t)NTOK * D_FF;       // 16M

    half_t* xh   = (half_t*)alloc(NE_X  * 2);
    half_t* Wqh  = (half_t*)alloc(NE_W  * 2);
    half_t* Wkh  = (half_t*)alloc(NE_W  * 2);
    half_t* Wvh  = (half_t*)alloc(NE_W  * 2);
    half_t* Woh  = (half_t*)alloc(NE_W  * 2);
    half_t* W1h  = (half_t*)alloc(NE_W1 * 2);
    half_t* W2h  = (half_t*)alloc(NE_W1 * 2);
    half_t* qh   = (half_t*)alloc(NE_X  * 2);
    half_t* kh   = (half_t*)alloc(NE_X  * 2);
    half_t* vh   = (half_t*)alloc(NE_X  * 2);
    half_t* oh   = (half_t*)alloc(NE_X  * 2);
    half_t* hh16 = (half_t*)alloc(NE_X  * 2);
    half_t* ffn1 = (half_t*)alloc(NE_F  * 2);
    float*  aprj = (float*)alloc(NE_X * 4);
    float*  hF   = (float*)alloc(NE_X * 4);
    float*  f2F  = (float*)alloc(NE_X * 4);

    // 1) fp32 -> f16 conversions
    cvt_f16_kernel<<<(int)(NE_X  / 256), 256, 0, stream>>>(x,  xh,  (int)NE_X);
    cvt_f16_kernel<<<(int)(NE_W  / 256), 256, 0, stream>>>(Wq, Wqh, (int)NE_W);
    cvt_f16_kernel<<<(int)(NE_W  / 256), 256, 0, stream>>>(Wk, Wkh, (int)NE_W);
    cvt_f16_kernel<<<(int)(NE_W  / 256), 256, 0, stream>>>(Wv, Wvh, (int)NE_W);
    cvt_f16_kernel<<<(int)(NE_W  / 256), 256, 0, stream>>>(Wo, Woh, (int)NE_W);
    cvt_f16_kernel<<<(int)(NE_W1 / 256), 256, 0, stream>>>(W1, W1h, (int)NE_W1);
    cvt_f16_kernel<<<(int)(NE_W1 / 256), 256, 0, stream>>>(W2, W2h, (int)NE_W1);

    // 2) QKV projections (f16 out)
    {
        dim3 grid(D_MODEL / TN, NTOK / TM);
        gemm_bias_kernel<<<grid, 256, 0, stream>>>(xh, Wqh, bq, nullptr, qh,
                                                   NTOK, D_MODEL, D_MODEL, 0);
        gemm_bias_kernel<<<grid, 256, 0, stream>>>(xh, Wkh, bk, nullptr, kh,
                                                   NTOK, D_MODEL, D_MODEL, 0);
        gemm_bias_kernel<<<grid, 256, 0, stream>>>(xh, Wvh, bv, nullptr, vh,
                                                   NTOK, D_MODEL, D_MODEL, 0);
    }

    // 3) Flash attention -> oh (f16)
    {
        dim3 grid(SEQ / 64, BATCH * NHEAD);
        flash_attn_kernel<<<grid, 128, 0, stream>>>(qh, kh, vh, oh);
    }

    // 4) Output projection (f32 out)
    {
        dim3 grid(D_MODEL / TN, NTOK / TM);
        gemm_bias_kernel<<<grid, 256, 0, stream>>>(oh, Woh, bo, aprj, nullptr,
                                                   NTOK, D_MODEL, D_MODEL, 0);
    }

    // 5) Residual + LN1 -> hF (f32) + hh16 (f16)
    residual_ln_kernel<<<NTOK, 256, 0, stream>>>(x, aprj, g1, be1, hF, hh16);

    // 6) FFN1 with ReLU (f16 out)
    {
        dim3 grid(D_FF / TN, NTOK / TM);
        gemm_bias_kernel<<<grid, 256, 0, stream>>>(hh16, W1h, b1, nullptr, ffn1,
                                                   NTOK, D_FF, D_MODEL, 1);
    }

    // 7) FFN2 (f32 out)
    {
        dim3 grid(D_MODEL / TN, NTOK / TM);
        gemm_bias_kernel<<<grid, 256, 0, stream>>>(ffn1, W2h, b2, f2F, nullptr,
                                                   NTOK, D_MODEL, D_FF, 0);
    }

    // 8) Residual + LN2 -> d_out (f32)
    residual_ln_kernel<<<NTOK, 256, 0, stream>>>(hF, f2F, g2, be2, out, nullptr);
}
